// SelfAttentionDeltaBlock_45715631899234
// MI455X (gfx1250) — compile-verified, matrix operand order checked by reading, not run on hardware
//
#include <hip/hip_runtime.h>
#include <stdint.h>
#include <stddef.h>

#define BB 2
#define SS 2048
#define DD 1024
#define HH 16
#define DHH 64
#define RR 8
#define MM (BB * SS)   // 4096 rows

typedef __bf16 bf16_t;
typedef __attribute__((ext_vector_type(16))) __bf16 v16bf;
typedef __attribute__((ext_vector_type(8)))  __bf16 v8bf;
typedef __attribute__((ext_vector_type(4)))  __bf16 v4bf;
typedef __attribute__((ext_vector_type(8)))  float  v8f;
typedef __attribute__((ext_vector_type(4)))  float  f32x4;
typedef __attribute__((ext_vector_type(4)))  unsigned int u32x4;
typedef __attribute__((ext_vector_type(4)))  int i32x4;
typedef __attribute__((ext_vector_type(8)))  int i32x8;

#if __has_builtin(__builtin_amdgcn_tensor_load_to_lds)
#define HAVE_TDM 1
#else
#define HAVE_TDM 0
#endif

// Load a 16-element bf16 WMMA fragment as two contiguous 16B chunks
// (matches CDNA5 16-bit A/B VGPR layout: per-lane K-halves are contiguous).
static __device__ inline v16bf frag_ld(const bf16_t* p0, const bf16_t* p1) {
  union { v16bf v; v8bf h[2]; } u;
  u.h[0] = *(const v8bf*)p0;
  u.h[1] = *(const v8bf*)p1;
  return u.v;
}

static __device__ inline v8f wmma_bf16(v16bf a, v16bf b, v8f c) {
  // (neg_a, A, neg_b, B, c_mod, C, reuse_a, reuse_b)
  return __builtin_amdgcn_wmma_f32_16x16x32_bf16(false, a, false, b, (short)0, c,
                                                 false, false);
}

#if HAVE_TDM
// Issue one TDM 2D tile load: global (row-major, strideElems between rows,
// 2-byte elements) -> LDS, packed contiguously (LDS row stride == tile_d0).
// D# packing per CDNA5 ISA. tensor_dim == tile_dim
// so no OOB clamping occurs. Tracked with TENSORcnt.
static __device__ inline void tdm_load_2d(uint32_t lds_off, const void* gptr,
                                          uint32_t tile_d0, uint32_t tile_d1,
                                          uint32_t strideElems) {
  const unsigned long long ga = (unsigned long long)(size_t)gptr;
  u32x4 g0;
  g0.x = 1u;                                            // count=1 (valid D#)
  g0.y = lds_off;                                       // lds_addr (bytes)
  g0.z = (uint32_t)(ga & 0xFFFFFFFFu);                  // global_addr[31:0]
  g0.w = (uint32_t)((ga >> 32) & 0x01FFFFFFu) | (2u << 30);  // addr[56:32]|type=2
  i32x8 g1;
  g1[0] = (int)(1u << 16);                              // data_size=1 -> 2 bytes
  g1[1] = (int)((tile_d0 & 0xFFFFu) << 16);             // tensor_dim0[15:0]
  g1[2] = (int)((tile_d0 >> 16) | ((tile_d1 & 0xFFFFu) << 16)); // d0 hi | d1 lo
  g1[3] = (int)((tile_d1 >> 16) | (tile_d0 << 16));     // d1 hi | tile_dim0
  g1[4] = (int)(tile_d1 & 0xFFFFu);                     // tile_dim1, tile_dim2=0
  g1[5] = (int)strideElems;                             // tensor_dim0_stride lo
  g1[6] = 0;                                            // stride hi / dim1 stride
  g1[7] = 0;
  const i32x4 z4 = {0, 0, 0, 0};
#if __clang_major__ >= 23
  const i32x8 z8 = {0, 0, 0, 0, 0, 0, 0, 0};
  __builtin_amdgcn_tensor_load_to_lds(g0, g1, z4, z4, z8, 0);
#else
  __builtin_amdgcn_tensor_load_to_lds(g0, g1, z4, z4, 0);
#endif
}
#endif  // HAVE_TDM

// ---------------------------------------------------------------------------
// RMSNorm (f32 in) -> bf16 out, one block (256 thr = 8 waves) per row.
// ---------------------------------------------------------------------------
__global__ __launch_bounds__(256) void k_rmsnorm(const float* __restrict__ x,
                                                 const float* __restrict__ w,
                                                 bf16_t* __restrict__ o) {
  const int row = blockIdx.x;
  const int tid = threadIdx.x;
  const float* xr = x + (size_t)row * DD;
  float xv[4];
  float ss = 0.f;
#pragma unroll
  for (int i = 0; i < 4; ++i) {
    xv[i] = xr[tid + i * 256];
    ss += xv[i] * xv[i];
  }
#pragma unroll
  for (int off = 16; off >= 1; off >>= 1) ss += __shfl_xor(ss, off, 32);
  __shared__ float red[8];
  if ((tid & 31) == 0) red[tid >> 5] = ss;
  __syncthreads();
  const float tot = red[0] + red[1] + red[2] + red[3] +
                    red[4] + red[5] + red[6] + red[7];
  const float scale = rsqrtf(tot * (1.0f / DD) + 1e-5f);
  bf16_t* orow = o + (size_t)row * DD;
#pragma unroll
  for (int i = 0; i < 4; ++i) {
    const int d = tid + i * 256;
    orow[d] = (bf16_t)(xv[i] * scale * w[d]);
  }
}

// ---------------------------------------------------------------------------
// f32 -> bf16 pack, 4 elements / thread.
// ---------------------------------------------------------------------------
__global__ __launch_bounds__(256) void k_f32_to_bf16(const float* __restrict__ s,
                                                     bf16_t* __restrict__ d, int n) {
  const int i = (blockIdx.x * 256 + threadIdx.x) * 4;
  if (i >= n) return;
  f32x4 v = *(const f32x4*)(s + i);
  v4bf o;
  o.x = (bf16_t)v.x; o.y = (bf16_t)v.y; o.z = (bf16_t)v.z; o.w = (bf16_t)v.w;
  *(v4bf*)(d + i) = o;
}

// ---------------------------------------------------------------------------
// C[M,N] = A[M,K](bf16) @ W[N,K](bf16)^T (+ optional f32 residual).
// Block tile 128x64, 8 waves -> wave tile 32x32 (2x2 WMMA 16x16), BK=32.
// Double-buffered LDS fed by TDM (tensor_load_to_lds, TENSORcnt) when
// available; VALU global->LDS fallback otherwise.
// ---------------------------------------------------------------------------
template <int K, int N, bool RES>
__global__ __launch_bounds__(256) void k_gemm_nt(const bf16_t* __restrict__ A,
                                                 const bf16_t* __restrict__ W,
                                                 const float* __restrict__ resid,
                                                 float* __restrict__ C) {
  __shared__ alignas(16) bf16_t As[2][128][32];
  __shared__ alignas(16) bf16_t Bs[2][64][32];
  const int tid = threadIdx.x;
  const int lane = tid & 31, wave = tid >> 5;
  const int mw = wave & 3, nw = wave >> 2;
  const int r16 = lane & 15, hl = lane >> 4;
  const int m0 = blockIdx.y * 128, n0 = blockIdx.x * 64;

  const v8f vzero = {};
  v8f acc[2][2];
#pragma unroll
  for (int i = 0; i < 2; ++i)
#pragma unroll
    for (int j = 0; j < 2; ++j) acc[i][j] = vzero;

#if HAVE_TDM
  // TDM path: wave 0 issues tile DMAs (EXEC-independent, one issue per wave);
  // everyone synchronizes after wave 0 drains TENSORcnt for the active buffer.
  if (wave == 0) {
    tdm_load_2d((uint32_t)(size_t)&As[0][0][0], A + (size_t)m0 * K, 32u, 128u, K);
    tdm_load_2d((uint32_t)(size_t)&Bs[0][0][0], W + (size_t)n0 * K, 32u, 64u, K);
  }
  int cur = 0;
  for (int kt = 0; kt < K / 32; ++kt) {
    const int nk = (kt + 1) * 32;
    if (nk < K) {
      if (wave == 0) {
        tdm_load_2d((uint32_t)(size_t)&As[cur ^ 1][0][0],
                    A + (size_t)m0 * K + nk, 32u, 128u, K);
        tdm_load_2d((uint32_t)(size_t)&Bs[cur ^ 1][0][0],
                    W + (size_t)n0 * K + nk, 32u, 64u, K);
      }
      __builtin_amdgcn_s_wait_tensorcnt((short)2);  // current buffer landed
    } else {
      __builtin_amdgcn_s_wait_tensorcnt((short)0);
    }
    __syncthreads();

    v16bf af[2], bw[2];
#pragma unroll
    for (int i = 0; i < 2; ++i) {
      const int r = mw * 32 + i * 16 + r16;
      af[i] = frag_ld(&As[cur][r][hl * 8], &As[cur][r][16 + hl * 8]);
    }
#pragma unroll
    for (int j = 0; j < 2; ++j) {
      const int r = nw * 32 + j * 16 + r16;
      bw[j] = frag_ld(&Bs[cur][r][hl * 8], &Bs[cur][r][16 + hl * 8]);
    }
#pragma unroll
    for (int i = 0; i < 2; ++i)
#pragma unroll
      for (int j = 0; j < 2; ++j)
        acc[i][j] = wmma_bf16(af[i], bw[j], acc[i][j]);
    __syncthreads();  // buffer free before next TDM overwrite
    cur ^= 1;
  }
#else
  const int ar = tid >> 1, ac = (tid & 1) * 16;
  const int br = tid >> 2, bc = (tid & 3) * 8;
  for (int k0 = 0; k0 < K; k0 += 32) {
    const bf16_t* asrc = A + (size_t)(m0 + ar) * K + k0 + ac;
    *(u32x4*)&As[0][ar][ac]     = *(const u32x4*)asrc;
    *(u32x4*)&As[0][ar][ac + 8] = *(const u32x4*)(asrc + 8);
    const bf16_t* bsrc = W + (size_t)(n0 + br) * K + k0 + bc;
    *(u32x4*)&Bs[0][br][bc] = *(const u32x4*)bsrc;
    __syncthreads();
    v16bf af[2], bw[2];
#pragma unroll
    for (int i = 0; i < 2; ++i) {
      const int r = mw * 32 + i * 16 + r16;
      af[i] = frag_ld(&As[0][r][hl * 8], &As[0][r][16 + hl * 8]);
    }
#pragma unroll
    for (int j = 0; j < 2; ++j) {
      const int r = nw * 32 + j * 16 + r16;
      bw[j] = frag_ld(&Bs[0][r][hl * 8], &Bs[0][r][16 + hl * 8]);
    }
#pragma unroll
    for (int i = 0; i < 2; ++i)
#pragma unroll
      for (int j = 0; j < 2; ++j)
        acc[i][j] = wmma_bf16(af[i], bw[j], acc[i][j]);
    __syncthreads();
  }
#endif

#pragma unroll
  for (int i = 0; i < 2; ++i)
#pragma unroll
    for (int j = 0; j < 2; ++j)
#pragma unroll
      for (int v = 0; v < 8; ++v) {
        const size_t row = (size_t)(m0 + mw * 32 + i * 16 + 8 * hl + v);
        const int col = n0 + nw * 32 + j * 16 + r16;
        float val = acc[i][j][v];
        if (RES) val += resid[row * N + col];
        C[row * N + col] = val;
      }
}

// ---------------------------------------------------------------------------
// Rank-8 LoRA delta on one row: o = bf16(h + 2 * (h @ A) @ Bp^T).
// grid = (MM, 2): y==0 -> Q (layers 0/2), y==1 -> K (layers 1/3).
// ---------------------------------------------------------------------------
__global__ __launch_bounds__(256) void k_lora(const float* __restrict__ hq,
                                              const float* __restrict__ hk,
                                              const float* __restrict__ lora,
                                              bf16_t* __restrict__ oq,
                                              bf16_t* __restrict__ ok) {
  const int row = blockIdx.x;
  const int which = blockIdx.y;
  const int tid = threadIdx.x;
  const int b = row / SS;
  const float* h = (which == 0 ? hq : hk) + (size_t)row * DD;
  bf16_t* o = (which == 0 ? oq : ok) + (size_t)row * DD;
  const float* Ap = lora + ((size_t)which * BB + b) * DD * RR;
  const float* Bp = lora + ((size_t)(which + 2) * BB + b) * DD * RR;

  __shared__ float zsh[RR];
  if (tid < RR) zsh[tid] = 0.f;
  __syncthreads();

  float zp[RR] = {};
  float hv4[4];
#pragma unroll
  for (int i = 0; i < 4; ++i) {
    const int d = tid + i * 256;
    const float hv = h[d];
    hv4[i] = hv;
#pragma unroll
    for (int r = 0; r < RR; ++r) zp[r] += hv * Ap[(size_t)d * RR + r];
  }
#pragma unroll
  for (int r = 0; r < RR; ++r) {
    float z = zp[r];
#pragma unroll
    for (int off = 16; off >= 1; off >>= 1) z += __shfl_xor(z, off, 32);
    if ((tid & 31) == 0) atomicAdd(&zsh[r], z);
  }
  __syncthreads();
  float zr[RR];
#pragma unroll
  for (int r = 0; r < RR; ++r) zr[r] = zsh[r];
#pragma unroll
  for (int i = 0; i < 4; ++i) {
    const int d = tid + i * 256;
    float delta = 0.f;
#pragma unroll
    for (int r = 0; r < RR; ++r) delta += zr[r] * Bp[(size_t)d * RR + r];
    o[d] = (bf16_t)(hv4[i] + 2.0f * delta);
  }
}

// ---------------------------------------------------------------------------
// Causal flash attention. Block = 128 thr (4 waves), 64 q-rows per block
// (16 rows/wave), KV tiles of 64. QK^T and P@V in bf16 WMMA, f32 accum.
// grid = (SS/64, BB*HH).
// ---------------------------------------------------------------------------
__global__ __launch_bounds__(128) void k_attn(const bf16_t* __restrict__ Q,
                                              const bf16_t* __restrict__ Km,
                                              const bf16_t* __restrict__ Vm,
                                              bf16_t* __restrict__ O) {
  __shared__ alignas(16) bf16_t Ks[64][64];     // [kv][dh]
  __shared__ alignas(16) bf16_t Vs[64][64];     // transposed: [dh][kv]
  __shared__ alignas(16) bf16_t Ps[4][16][64];  // per-wave P staging

  const int tid = threadIdx.x;
  const int lane = tid & 31, wave = tid >> 5;
  const int r16 = lane & 15, hl = lane >> 4;
  const int qblk = blockIdx.x;
  const int b = blockIdx.y >> 4, h = blockIdx.y & 15;  // HH == 16
  const int qbase = qblk * 64;

  // Q fragments stay resident (A-layout: lane -> row, two K-steps of 32).
  v16bf aq[2];
  {
    const bf16_t* qrow =
        Q + ((size_t)(b * SS + qbase + wave * 16 + r16)) * DD + h * DHH;
#pragma unroll
    for (int kk = 0; kk < 2; ++kk)
      aq[kk] = frag_ld(qrow + kk * 32 + hl * 8, qrow + kk * 32 + 16 + hl * 8);
  }

  const v8f vzero = {};
  float mrun[8], lrun[8];
  v8f oacc[4];
#pragma unroll
  for (int f = 0; f < 4; ++f) oacc[f] = vzero;
#pragma unroll
  for (int v = 0; v < 8; ++v) { mrun[v] = -1e30f; lrun[v] = 0.f; }

  for (int j = 0; j <= qblk; ++j) {
    const int jb = j * 64;
    {
      const int kv = tid & 63, ch = (tid >> 6) * 32;
      const bf16_t* ksrc = Km + (size_t)(b * SS + jb + kv) * DD + h * DHH + ch;
#pragma unroll
      for (int u = 0; u < 4; ++u)
        *(u32x4*)&Ks[kv][ch + u * 8] = *(const u32x4*)(ksrc + u * 8);
      const bf16_t* vsrc = Vm + (size_t)(b * SS + jb + kv) * DD + h * DHH + ch;
#pragma unroll
      for (int u = 0; u < 32; ++u) Vs[ch + u][kv] = vsrc[u];  // transpose
      if (j < qblk) {  // prefetch next KV tile rows (-> global_prefetch_b8)
        __builtin_prefetch(ksrc + (size_t)64 * DD, 0, 1);
        __builtin_prefetch(vsrc + (size_t)64 * DD, 0, 1);
      }
    }
    __syncthreads();

    // S = Q @ K^T  (B-matrix column n -> K-tile row n, contiguous dh)
    v8f sf[4];
#pragma unroll
    for (int f = 0; f < 4; ++f) {
      v8f c = vzero;
#pragma unroll
      for (int kk = 0; kk < 2; ++kk) {
        const int n = f * 16 + r16;
        v16bf bk = frag_ld(&Ks[n][kk * 32 + hl * 8],
                           &Ks[n][kk * 32 + 16 + hl * 8]);
        c = wmma_bf16(aq[kk], bk, c);
      }
      sf[f] = c;
    }

    // scale + causal mask (C-layout: row = v + 8*hl, col = f*16 + r16)
#pragma unroll
    for (int f = 0; f < 4; ++f)
#pragma unroll
      for (int v = 0; v < 8; ++v) {
        const int qi = qbase + wave * 16 + 8 * hl + v;
        const int ki = jb + f * 16 + r16;
        const float s = sf[f][v] * 0.125f;  // 1/sqrt(64)
        sf[f][v] = (ki <= qi) ? s : -1e30f;
      }

    // online softmax: row stats live in one 16-lane half -> width-16 shuffles
    float mnew[8], alpha[8];
#pragma unroll
    for (int v = 0; v < 8; ++v) {
      float mx = fmaxf(fmaxf(sf[0][v], sf[1][v]), fmaxf(sf[2][v], sf[3][v]));
#pragma unroll
      for (int off = 8; off >= 1; off >>= 1) mx = fmaxf(mx, __shfl_xor(mx, off, 16));
      mnew[v] = fmaxf(mrun[v], mx);
      alpha[v] = __expf(mrun[v] - mnew[v]);
      mrun[v] = mnew[v];
    }
#pragma unroll
    for (int f = 0; f < 4; ++f)
#pragma unroll
      for (int v = 0; v < 8; ++v) sf[f][v] = __expf(sf[f][v] - mnew[v]);
#pragma unroll
    for (int v = 0; v < 8; ++v) {
      float rs = sf[0][v] + sf[1][v] + sf[2][v] + sf[3][v];
#pragma unroll
      for (int off = 8; off >= 1; off >>= 1) rs += __shfl_xor(rs, off, 16);
      lrun[v] = lrun[v] * alpha[v] + rs;
    }

    // stage P through per-wave LDS (C-layout -> A-layout re-read, same wave:
    // DS ops are in-order + compiler-inserted s_wait_dscnt)
#pragma unroll
    for (int f = 0; f < 4; ++f)
#pragma unroll
      for (int v = 0; v < 8; ++v)
        Ps[wave][v + 8 * hl][f * 16 + r16] = (bf16_t)sf[f][v];

    v16bf ap[2];
    {
      const bf16_t* prow = &Ps[wave][r16][0];
#pragma unroll
      for (int kk = 0; kk < 2; ++kk)
        ap[kk] = frag_ld(prow + kk * 32 + hl * 8, prow + kk * 32 + 16 + hl * 8);
    }

    // O = O*alpha + P @ V  (V transposed so B-frags load contiguous kv)
#pragma unroll
    for (int f = 0; f < 4; ++f) {
#pragma unroll
      for (int v = 0; v < 8; ++v) oacc[f][v] *= alpha[v];
#pragma unroll
      for (int kk = 0; kk < 2; ++kk) {
        const int n = f * 16 + r16;
        v16bf bv = frag_ld(&Vs[n][kk * 32 + hl * 8],
                           &Vs[n][kk * 32 + 16 + hl * 8]);
        oacc[f] = wmma_bf16(ap[kk], bv, oacc[f]);
      }
    }
    __syncthreads();
  }

  float inv[8];
#pragma unroll
  for (int v = 0; v < 8; ++v) inv[v] = 1.0f / lrun[v];
#pragma unroll
  for (int f = 0; f < 4; ++f)
#pragma unroll
    for (int v = 0; v < 8; ++v) {
      const size_t row = (size_t)(b * SS + qbase + wave * 16 + 8 * hl + v);
      const int col = h * DHH + f * 16 + r16;
      O[row * DD + col] = (bf16_t)(oacc[f][v] * inv[v]);
    }
}

// ---------------------------------------------------------------------------
// Host launcher
// ---------------------------------------------------------------------------
extern "C" void kernel_launch(void* const* d_in, const int* in_sizes, int n_in,
                              void* d_out, int out_size, void* d_ws, size_t ws_size,
                              hipStream_t stream) {
  (void)in_sizes; (void)n_in; (void)out_size; (void)ws_size;
  const float* hidden = (const float*)d_in[0];
  // d_in[1] = attention_mask (causal, recomputed in-kernel)
  const float* lora = (const float*)d_in[2];
  const float* ln_w = (const float*)d_in[3];
  const float* Wq = (const float*)d_in[4];
  const float* Wk = (const float*)d_in[5];
  const float* Wv = (const float*)d_in[6];
  const float* Wo = (const float*)d_in[7];
  float* out = (float*)d_out;

  char* ws = (char*)d_ws;
  const size_t MB = 1ull << 20;
  bf16_t* xln = (bf16_t*)(ws + 0 * MB);   // 8 MiB  : LN(x) bf16
  float*  hq  = (float*) (ws + 8 * MB);   // 16 MiB : x@Wq^T f32
  float*  hk  = (float*) (ws + 24 * MB);  // 16 MiB
  float*  hv  = (float*) (ws + 40 * MB);  // 16 MiB
  bf16_t* qb  = (bf16_t*)(ws + 56 * MB);  // 8 MiB  : q (post-LoRA) bf16
  bf16_t* kb  = (bf16_t*)(ws + 64 * MB);  // 8 MiB
  bf16_t* vb  = (bf16_t*)(ws + 72 * MB);  // 8 MiB
  bf16_t* ob  = (bf16_t*)(ws + 80 * MB);  // 8 MiB  : attention output bf16
  bf16_t* wqb = (bf16_t*)(ws + 88 * MB);  // 2 MiB each: bf16 weights
  bf16_t* wkb = (bf16_t*)(ws + 90 * MB);
  bf16_t* wvb = (bf16_t*)(ws + 92 * MB);
  bf16_t* wob = (bf16_t*)(ws + 94 * MB);

  k_rmsnorm<<<MM, 256, 0, stream>>>(hidden, ln_w, xln);
  k_f32_to_bf16<<<1024, 256, 0, stream>>>(Wq, wqb, DD * DD);
  k_f32_to_bf16<<<1024, 256, 0, stream>>>(Wk, wkb, DD * DD);
  k_f32_to_bf16<<<1024, 256, 0, stream>>>(Wv, wvb, DD * DD);
  k_f32_to_bf16<<<1024, 256, 0, stream>>>(Wo, wob, DD * DD);

  dim3 gg(DD / 64, MM / 128);  // (16, 32)
  k_gemm_nt<DD, DD, false><<<gg, 256, 0, stream>>>(xln, wqb, nullptr, hq);
  k_gemm_nt<DD, DD, false><<<gg, 256, 0, stream>>>(xln, wkb, nullptr, hk);
  k_gemm_nt<DD, DD, false><<<gg, 256, 0, stream>>>(xln, wvb, nullptr, hv);

  k_lora<<<dim3(MM, 2), 256, 0, stream>>>(hq, hk, lora, qb, kb);
  k_f32_to_bf16<<<(MM * DD) / 1024, 256, 0, stream>>>(hv, vb, MM * DD);

  k_attn<<<dim3(SS / 64, BB * HH), 128, 0, stream>>>(qb, kb, vb, ob);

  k_gemm_nt<DD, DD, true><<<gg, 256, 0, stream>>>(ob, wob, hidden, out);
}
